// TransitionDown_v2_48095043780747
// MI455X (gfx1250) — compile-verified, matrix operand order checked by reading, not run on hardware
//
#include <hip/hip_runtime.h>

#define CIN   256
#define COUT  512
#define KNN   16
#define MB    4            // sampled points per workgroup (weight-reuse blocking)
#define LDSS  264          // bf16 elements per A row (256 + 8 pad = 16B-aligned stride)
#define NWAVE 8
#define BROW  40           // bf16 elements per staged B row (32 + 4DW pad from TDM)

typedef __attribute__((ext_vector_type(16))) __bf16 v16bf;
typedef __attribute__((ext_vector_type(8)))  __bf16 v8bf;
typedef __attribute__((ext_vector_type(8)))  float  v8f;
typedef __attribute__((ext_vector_type(4)))  unsigned int u32x4;
typedef __attribute__((ext_vector_type(8)))  int i32x8;
typedef __attribute__((ext_vector_type(4)))  int i32x4;

__device__ __forceinline__ v16bf cat8(v8bf lo, v8bf hi) {
  return __builtin_shufflevector(lo, hi, 0,1,2,3,4,5,6,7,8,9,10,11,12,13,14,15);
}

// ---------------------------------------------------------------------------
// TDM: async DMA of a 64(cout) x 32(k) bf16 weight tile into LDS.
// 2D descriptor per CDNA5 ISA; LDS rows padded 64B->80B
// (pad_interval = 16 DW, pad_amount = 4 DW) for bank-conflict-free ds_load_b128.
// ---------------------------------------------------------------------------
__device__ __forceinline__ void tdm_load_wtile(const __bf16* gaddr, unsigned lds_off) {
  unsigned long long ga = (unsigned long long)(size_t)gaddr;
  u32x4 g0;
  g0[0] = 1u;                                   // count=1, user descriptor
  g0[1] = lds_off;                              // lds_addr (bytes)
  g0[2] = (unsigned)ga;                         // global_addr[31:0]
  g0[3] = (unsigned)(ga >> 32) | (2u << 30);    // global_addr[56:32] | type=2

  i32x8 g1;
  g1[0] = (int)((1u << 16) | (1u << 20) | (3u << 22) | (3u << 25));
        // data_size=2B | pad_enable | pad_interval=16DW | pad_amount=4DW
  g1[1] = (int)(((unsigned)CIN & 0xFFFFu) << 16);   // tensor_dim0 = 256 (lo16)
  g1[2] = (int)(64u << 16);                         // tensor_dim1 = 64 rows from tile start
  g1[3] = (int)(32u << 16);                         // tile_dim0 = 32 (k)
  g1[4] = (int)(64u);                               // tile_dim1 = 64 (couts), tile_dim2 = 0
  g1[5] = (int)(CIN);                               // tensor_dim0_stride = 256 elements
  g1[6] = 0;
  g1[7] = 0;

  i32x4 z4 = {0, 0, 0, 0};                          // 2D: groups 2/3 unused
  i32x8 z8 = {0, 0, 0, 0, 0, 0, 0, 0};
  __builtin_amdgcn_tensor_load_to_lds(g0, g1, z4, z4, z8, 0);
}

// ---------------------------------------------------------------------------
// Main fused kernel: gather -> LayerNorm -> (TDM-fed) bf16 WMMA -> max over K
// grid.x = ceil(M / MB), block = 256 threads (8 wave32)
// ---------------------------------------------------------------------------
__global__ __launch_bounds__(256)
void td_main_kernel(const float* __restrict__ feats,
                    const float* __restrict__ norm_w,
                    const float* __restrict__ norm_b,
                    const __bf16* __restrict__ Wb,     // [COUT][CIN] bf16
                    const int* __restrict__ knn,       // [M][KNN]
                    float* __restrict__ outFeat,       // [M][COUT]
                    int M)
{
  __shared__ __attribute__((aligned(16))) __bf16 As[MB * KNN * LDSS];
  __shared__ __attribute__((aligned(16))) __bf16 Bs[2][NWAVE][64 * BROW];

  const int tid = threadIdx.x;
  const int m0  = blockIdx.x * MB;
  const int wv  = __builtin_amdgcn_readfirstlane((int)(tid >> 5));  // wave id (SGPR)
  const int lane = tid & 31;
  const int nsub = lane & 15;
  const int hi   = lane >> 4;          // 0: lanes 0-15, 1: lanes 16-31

  // ---- kick off the first two async weight tiles; DMA overlaps phase 1 ----
  const __bf16* wbase = Wb + (size_t)wv * 64 * CIN;   // this wave's 64 couts
  const unsigned ldsb0 = (unsigned)(size_t)&Bs[0][wv][0];
  const unsigned ldsb1 = (unsigned)(size_t)&Bs[1][wv][0];
  tdm_load_wtile(wbase + 0,  ldsb0);   // k-step 0
  tdm_load_wtile(wbase + 32, ldsb1);   // k-step 1

  // ---------- Phase 1: gather neighbor features + LayerNorm -> bf16 LDS ----------
  {
    const int row = tid >> 2;            // 0..63  == mb*16 + rr
    const int mb  = row >> 4;
    const int rr  = row & 15;
    const int c0  = (tid & 3) * 64;      // 4 threads cover one 256-channel row

    int mEff = m0 + mb; if (mEff > M - 1) mEff = M - 1;   // clamp tail block
    const int nidx = knn[mEff * KNN + rr];
    const float4* fr = (const float4*)(feats + (size_t)nidx * CIN + c0);

    float xs[64];
    float s = 0.f, ss = 0.f;
#pragma unroll
    for (int v = 0; v < 16; ++v) {
      float4 f4 = fr[v];
      xs[4*v+0] = f4.x; xs[4*v+1] = f4.y; xs[4*v+2] = f4.z; xs[4*v+3] = f4.w;
    }
#pragma unroll
    for (int j = 0; j < 64; ++j) { s += xs[j]; ss += xs[j] * xs[j]; }
    s += __shfl_xor(s, 1, 32);  ss += __shfl_xor(ss, 1, 32);
    s += __shfl_xor(s, 2, 32);  ss += __shfl_xor(ss, 2, 32);

    const float mean = s * (1.0f / CIN);
    const float var  = ss * (1.0f / CIN) - mean * mean;
    const float rsv  = rsqrtf(var + 1e-5f);

    __bf16* dst = As + row * LDSS + c0;
#pragma unroll
    for (int j = 0; j < 64; ++j) {
      float nv = (xs[j] - mean) * rsv * norm_w[c0 + j] + norm_b[c0 + j];
      dst[j] = (__bf16)nv;
    }
  }
  __syncthreads();

  // ---------- Phase 2: [16(k) x 256] x [256 x 512] per m, TDM-fed WMMA ----------
  v8f acc[MB][4];
#pragma unroll
  for (int mb = 0; mb < MB; ++mb)
#pragma unroll
    for (int t = 0; t < 4; ++t)
      acc[mb][t] = (v8f){};

#pragma unroll
  for (int kk8 = 0; kk8 < 8; ++kk8) {
    const int kk  = kk8 * 32;
    const int buf = kk8 & 1;

    // older of the outstanding TDM loads (this buffer) has landed
    if (kk8 == 7) {
      __builtin_amdgcn_s_wait_tensorcnt(0);
    } else {
      __builtin_amdgcn_s_wait_tensorcnt(1);
    }

    // A fragments (16-bit A 16x32 lane layout: low half K{0..7,16..23},
    // high half K{8..15,24..31}); two ds_load_b128 per fragment
    v16bf a[MB];
#pragma unroll
    for (int mb = 0; mb < MB; ++mb) {
      const __bf16* ap = As + (mb * KNN + nsub) * LDSS + kk + hi * 8;
      v8bf alo = *(const v8bf*)ap;
      v8bf ahi = *(const v8bf*)(ap + 16);
      a[mb] = cat8(alo, ahi);
    }
#pragma unroll
    for (int t = 0; t < 4; ++t) {
      const __bf16* bp = &Bs[buf][wv][(t * 16 + nsub) * BROW + hi * 8];
      v8bf blo = *(const v8bf*)bp;        // ds_load_b128, conflict-free (80B stride)
      v8bf bhi = *(const v8bf*)(bp + 16);
      v16bf b = cat8(blo, bhi);
#pragma unroll
      for (int mb = 0; mb < MB; ++mb) {
        acc[mb][t] = __builtin_amdgcn_wmma_f32_16x16x32_bf16(
            false, a[mb], false, b, (short)0, acc[mb][t], false, false);
      }
    }

    // refill this buffer with k-step kk8+2 once our LDS reads have retired
    if (kk8 + 2 < 8) {
      asm volatile("s_wait_dscnt 0x0" ::: "memory");
      tdm_load_wtile(wbase + (kk + 64), buf ? ldsb1 : ldsb0);
    }
  }

  // ---------- Max over the 16 neighbors straight from the C layout ----------
  // lane L (L<16): VGPR v holds (M=v, N=L); lane L+16: (M=8+v, N=L)
#pragma unroll
  for (int mb = 0; mb < MB; ++mb) {
    const int m = m0 + mb;
#pragma unroll
    for (int t = 0; t < 4; ++t) {
      v8f v = acc[mb][t];
      float mx = v[0];
#pragma unroll
      for (int e = 1; e < 8; ++e) mx = fmaxf(mx, v[e]);
      mx = fmaxf(mx, __shfl_xor(mx, 16, 32));     // rows 0-7 vs 8-15
      if (m < M && hi == 0) {
        const int n0 = (wv * 4 + t) * 16;
        outFeat[(size_t)m * COUT + n0 + nsub] = mx;
      }
    }
  }
}

// ---------------------------------------------------------------------------
// Weights f32 -> bf16 into workspace (runs once per launch, L2-resident after)
// ---------------------------------------------------------------------------
__global__ __launch_bounds__(256)
void td_wcvt_kernel(const float* __restrict__ w, __bf16* __restrict__ wb, int n) {
  int i = blockIdx.x * blockDim.x + threadIdx.x;
  if (i < n) wb[i] = (__bf16)w[i];
}

// ---------------------------------------------------------------------------
// n_xyz gather + n_offset scalar (int bits into the flat f32 output buffer)
// ---------------------------------------------------------------------------
__global__ __launch_bounds__(256)
void td_xyz_kernel(const float* __restrict__ xyz, const int* __restrict__ samp,
                   float* __restrict__ out_base, int M) {
  int i = blockIdx.x * blockDim.x + threadIdx.x;
  if (i < M) {
    int s3 = samp[i] * 3;
    out_base[(size_t)i * 3 + 0] = xyz[s3 + 0];
    out_base[(size_t)i * 3 + 1] = xyz[s3 + 1];
    out_base[(size_t)i * 3 + 2] = xyz[s3 + 2];
  }
  if (i == 0) {
    ((int*)out_base)[(size_t)M * 3 + (size_t)M * COUT] = M;   // n_offset
  }
}

// ---------------------------------------------------------------------------
extern "C" void kernel_launch(void* const* d_in, const int* in_sizes, int n_in,
                              void* d_out, int out_size, void* d_ws, size_t ws_size,
                              hipStream_t stream) {
  const float* xyz    = (const float*)d_in[0];
  const float* feats  = (const float*)d_in[1];
  const float* norm_w = (const float*)d_in[2];
  const float* norm_b = (const float*)d_in[3];
  const float* lin_w  = (const float*)d_in[4];
  const int*   samp   = (const int*)d_in[5];
  const int*   knn    = (const int*)d_in[6];
  const int M = in_sizes[5];                       // 16385 sampled points

  __bf16* Wb = (__bf16*)d_ws;                      // 512*256*2 = 256 KB scratch

  td_wcvt_kernel<<<(COUT * CIN + 255) / 256, 256, 0, stream>>>(lin_w, Wb, COUT * CIN);
  td_xyz_kernel<<<(M + 255) / 256, 256, 0, stream>>>(xyz, samp, (float*)d_out, M);

  float* outFeat = (float*)d_out + (size_t)M * 3;  // tuple order: n_xyz | out | n_offset
  td_main_kernel<<<(M + MB - 1) / MB, 256, 0, stream>>>(feats, norm_w, norm_b,
                                                        Wb, knn, outFeat, M);
}
